// MemoryBank_51135880626820
// MI455X (gfx1250) — matrix operand order repaired and checked by reading, not compiled, not run on hardware
//
#include <hip/hip_runtime.h>
#include <hip/hip_bf16.h>
#include <math.h>

// ---------------------------------------------------------------------------
// Fused cosine-sim retrieval:
//   sim = l2norm(Q) @ l2norm(Mem)^T  (f16 WMMA, f32 accum)
//   kernel2: 4-wave blocks share an LDS-staged, double-buffered B tile;
//   each wave computes 32 query rows (2 WMMAs per k-step) -> 8x less L2
//   traffic than wave-private streaming.  Streaming top-8 with per-row LDS
//   thresholds + branchless min/max filter.  Then softmax + gather + rescale.
// ---------------------------------------------------------------------------

#define D_DIM   256
#define KSTEP   32        // WMMA K per instruction (f16)
#define NCHUNK  16        // M-dimension split for occupancy
#define TOPK    8
#define WAVES   4         // waves per block in kernel 2
#define QROWS_PER_WAVE 32
#define QROWS_PER_BLOCK (WAVES * QROWS_PER_WAVE)   // 128
#define SELF_MATCH 0.9999f
#define EPS 1e-12f

typedef _Float16 v16h __attribute__((ext_vector_type(16)));
typedef _Float16 v8h  __attribute__((ext_vector_type(8)));
typedef float    v8f  __attribute__((ext_vector_type(8)));

// ------------------------- kernel 1: row normalize -------------------------
__global__ __launch_bounds__(256) void normalize_rows_kernel(
    const float* __restrict__ src, _Float16* __restrict__ dst,
    float* __restrict__ norms, int rows)
{
    const int lane = threadIdx.x & 31;
    const int row  = blockIdx.x * (blockDim.x >> 5) + (threadIdx.x >> 5);
    if (row >= rows) return;

    const float* p = src + (size_t)row * D_DIM + lane * 8;
    float4 a = *(const float4*)(p);
    float4 b = *(const float4*)(p + 4);
    float x[8] = {a.x, a.y, a.z, a.w, b.x, b.y, b.z, b.w};

    float ss = 0.f;
#pragma unroll
    for (int t = 0; t < 8; ++t) ss += x[t] * x[t];
#pragma unroll
    for (int off = 16; off >= 1; off >>= 1) ss += __shfl_xor(ss, off, 32);

    float n = sqrtf(ss);
    if (norms && lane == 0) norms[row] = n;
    float inv = 1.0f / fmaxf(n, EPS);

    v8h h;
#pragma unroll
    for (int t = 0; t < 8; ++t) h[t] = (_Float16)(x[t] * inv);
    *(v8h*)(dst + (size_t)row * D_DIM + lane * 8) = h;
}

// --------------------- kernel 2: fused GEMM + top-k ------------------------
// Block = 4 waves, 128 query rows, one M-chunk.  B tile (16 mem rows x 256
// dims, f16, 8KB) staged in LDS, double-buffered: each thread copies 4x16B
// (global_load_b128 -> ds_store_b128), one __syncthreads per tile.
// Each wave: 32 query rows = A fragments aA (rows 0-15) + aB (rows 16-31),
// 16 WMMAs per tile over two independent accumulator chains.
// Top-k: lane L owns query row (wavebase + L); per-row thresholds live in
// per-wave LDS; branchless filter (min(S-c, c-t) > 0) + one __any per tile.

__global__ __launch_bounds__(WAVES * 32) void simtopk_kernel(
    const _Float16* __restrict__ nq, const _Float16* __restrict__ nm,
    float* __restrict__ pvals, int* __restrict__ pidx, int chunkRows)
{
    __shared__ __align__(16) _Float16 bstage[2][16 * D_DIM];   // 2 x 8KB
    __shared__ __align__(16) float scratch[WAVES][32 * 16];    // sim tiles
    __shared__ __align__(16) float thr[WAVES][32];             // row thresholds

    const int tid  = threadIdx.x;
    const int lane = tid & 31;
    const int w    = tid >> 5;
    const int ci     = blockIdx.y;            // M-chunk index
    const int n0base = ci * chunkRows;
    const int qbase  = blockIdx.x * QROWS_PER_BLOCK + w * QROWS_PER_WAVE;

    const int frow  = lane & 15;              // A row / B col within tile
    const int col   = lane & 15;              // C column
    const int rbase = (lane >> 4) << 3;       // C row base per lane-half
    const int akb   = (lane >> 4) << 4;       // A lane-half K base: 0 or 16
    const int bkb   = (lane >> 4) << 4;       // B lane-half K base: 0 or 16

    thr[w][lane] = -__builtin_inff();

    // Preload A fragments: rows 0-15 (aA) and 16-31 (aB) of this wave's tile.
    v16h aA[8], aB[8];
    {
        const _Float16* qrowA = nq + (size_t)(qbase + frow) * D_DIM;
        const _Float16* qrowB = qrowA + (size_t)16 * D_DIM;
#pragma unroll
        for (int ks = 0; ks < 8; ++ks) {
            union { v16h v; v8h h[2]; } u;
            u.h[0] = *(const v8h*)(qrowA + ks * KSTEP + akb);
            u.h[1] = *(const v8h*)(qrowA + ks * KSTEP + akb + 8);
            aA[ks] = u.v;
            u.h[0] = *(const v8h*)(qrowB + ks * KSTEP + akb);
            u.h[1] = *(const v8h*)(qrowB + ks * KSTEP + akb + 8);
            aB[ks] = u.v;
        }
    }

    // Per-lane sorted-ascending top-8 for query row qbase+lane.
    float vals[TOPK];
    int   idxs[TOPK];
#pragma unroll
    for (int t = 0; t < TOPK; ++t) { vals[t] = -__builtin_inff(); idxs[t] = 0; }

    const int ntiles = chunkRows >> 4;

    // Stage tile 0 into LDS buffer 0.
    uint4 stg[4];
    {
        const uint4* gp = (const uint4*)(nm + (size_t)n0base * D_DIM);
#pragma unroll
        for (int i = 0; i < 4; ++i) stg[i] = gp[tid + 128 * i];
        uint4* sp = (uint4*)bstage[0];
#pragma unroll
        for (int i = 0; i < 4; ++i) sp[tid + 128 * i] = stg[i];
    }
    __syncthreads();

    for (int tI = 0; tI < ntiles; ++tI) {
        const int st = tI & 1;
        const int n0 = n0base + tI * 16;

        // Issue global loads for the NEXT tile (latency hidden by compute).
        // Final iteration over-reads one tile past the chunk: stays inside
        // the workspace allocation (regions following nm), never consumed.
        {
            const uint4* gp =
                (const uint4*)(nm + (size_t)(n0 + 16) * D_DIM);
#pragma unroll
            for (int i = 0; i < 4; ++i) stg[i] = gp[tid + 128 * i];
        }

        // B fragments from LDS (one cluster of 16 ds_load_b128).
        v16h bfrag[8];
        {
            const _Float16* brow = &bstage[st][frow * D_DIM + bkb];
#pragma unroll
            for (int ks = 0; ks < 8; ++ks)
                bfrag[ks] = *(const v16h*)(brow + ks * KSTEP);
        }

        // 16 WMMAs over two independent accumulator chains.
        v8f cA = {}, cB = {};
#pragma unroll
        for (int ks = 0; ks < 8; ++ks) {
            cA = __builtin_amdgcn_wmma_f32_16x16x32_f16(
                    false, aA[ks], false, bfrag[ks], (short)0, cA, false, false);
            cB = __builtin_amdgcn_wmma_f32_16x16x32_f16(
                    false, aB[ks], false, bfrag[ks], (short)0, cB, false, false);
        }

        // Branchless threshold filter:  t < c < S  <=>  min(S-c, c-t) > 0.
        float4 tA0 = *(const float4*)&thr[w][rbase];
        float4 tA1 = *(const float4*)&thr[w][rbase + 4];
        float4 tB0 = *(const float4*)&thr[w][16 + rbase];
        float4 tB1 = *(const float4*)&thr[w][16 + rbase + 4];
        float ttA[8] = {tA0.x, tA0.y, tA0.z, tA0.w, tA1.x, tA1.y, tA1.z, tA1.w};
        float ttB[8] = {tB0.x, tB0.y, tB0.z, tB0.w, tB1.x, tB1.y, tB1.z, tB1.w};
        float h = -1.0f;
#pragma unroll
        for (int r = 0; r < 8; ++r) {
            h = fmaxf(h, fminf(SELF_MATCH - cA[r], cA[r] - ttA[r]));
            h = fmaxf(h, fminf(SELF_MATCH - cB[r], cB[r] - ttB[r]));
        }

        if (__any(h > 0.0f)) {        // wave-uniform; intra-wave LDS only
#pragma unroll
            for (int r = 0; r < 8; ++r) {
                scratch[w][(r + rbase) * 16 + col]      = cA[r];
                scratch[w][(16 + r + rbase) * 16 + col] = cB[r];
            }
            // wave-private LDS: DS ops are in-order, no barrier needed
            float4 t0 = *(const float4*)&scratch[w][lane * 16 + 0];
            float4 t1 = *(const float4*)&scratch[w][lane * 16 + 4];
            float4 t2 = *(const float4*)&scratch[w][lane * 16 + 8];
            float4 t3 = *(const float4*)&scratch[w][lane * 16 + 12];
            float tv[16] = {t0.x, t0.y, t0.z, t0.w, t1.x, t1.y, t1.z, t1.w,
                            t2.x, t2.y, t2.z, t2.w, t3.x, t3.y, t3.z, t3.w};
#pragma unroll
            for (int j = 0; j < 16; ++j) {
                float v = tv[j];
                if (v > vals[0] && v < SELF_MATCH) {
                    vals[0] = v; idxs[0] = n0 + j;
#pragma unroll
                    for (int t = 0; t < TOPK - 1; ++t) {
                        if (vals[t] > vals[t + 1]) {
                            float tv2 = vals[t];
                            vals[t] = vals[t + 1]; vals[t + 1] = tv2;
                            int ti = idxs[t];
                            idxs[t] = idxs[t + 1]; idxs[t + 1] = ti;
                        }
                    }
                }
            }
            thr[w][lane] = vals[0];
        }

        // Store the prefetched tile into the other LDS buffer; hand off.
        {
            uint4* sp = (uint4*)bstage[st ^ 1];
#pragma unroll
            for (int i = 0; i < 4; ++i) sp[tid + 128 * i] = stg[i];
        }
        __syncthreads();
    }

    // Every lane owns one query row.
    {
        const size_t q = (size_t)(qbase + lane);
        float* pv = pvals + (q * NCHUNK + ci) * TOPK;
        int*   pi = pidx  + (q * NCHUNK + ci) * TOPK;
#pragma unroll
        for (int t = 0; t < TOPK; ++t) { pv[t] = vals[t]; pi[t] = idxs[t]; }
    }
}

// -------------------- kernel 3: merge + softmax + gather -------------------
// One wave per query: merge NCHUNK*TOPK = 128 candidates (4 per lane) via 8
// shfl-argmax rounds, softmax, weighted sum of raw f32 rows, rescale.
__global__ __launch_bounds__(256) void finalize_kernel(
    const float* __restrict__ pvals, const int* __restrict__ pidx,
    const float* __restrict__ memory, const float* __restrict__ qnorm,
    float* __restrict__ out, int Bq)
{
    const int lane = threadIdx.x & 31;
    const int q    = blockIdx.x * (blockDim.x >> 5) + (threadIdx.x >> 5);
    if (q >= Bq) return;

    const float* pv = pvals + (size_t)q * (NCHUNK * TOPK);
    const int*   pi = pidx  + (size_t)q * (NCHUNK * TOPK);
    float cv[4]; int cidx[4];
#pragma unroll
    for (int s = 0; s < 4; ++s) {
        cv[s]   = pv[lane + 32 * s];
        cidx[s] = pi[lane + 32 * s];
    }

    float topv[TOPK];
    int   topi[TOPK];
#pragma unroll
    for (int t = 0; t < TOPK; ++t) {
        float mv = cv[0]; int ms = 0;
#pragma unroll
        for (int s = 1; s < 4; ++s) if (cv[s] > mv) { mv = cv[s]; ms = s; }
        int mi = cidx[ms];
        int mo = lane * 4 + ms;
#pragma unroll
        for (int off = 16; off >= 1; off >>= 1) {
            float ov = __shfl_xor(mv, off, 32);
            int   oi = __shfl_xor(mi, off, 32);
            int   oo = __shfl_xor(mo, off, 32);
            if (ov > mv || (ov == mv && oo < mo)) { mv = ov; mi = oi; mo = oo; }
        }
        topv[t] = mv; topi[t] = mi;
        if ((mo >> 2) == lane) cv[mo & 3] = -__builtin_inff();
    }

    // softmax over 8 (topv[0] is the max)
    float w[TOPK], wsum = 0.f;
#pragma unroll
    for (int t = 0; t < TOPK; ++t) { w[t] = __expf(topv[t] - topv[0]); wsum += w[t]; }
    float winv = 1.0f / wsum;

    // gather + weighted sum: lane owns dims [lane*8, lane*8+8)
    float acc[8];
#pragma unroll
    for (int d = 0; d < 8; ++d) acc[d] = 0.f;
#pragma unroll
    for (int t = 0; t < TOPK; ++t) {
        const float* row = memory + (size_t)topi[t] * D_DIM + lane * 8;
        float wt = w[t] * winv;
        float4 a = *(const float4*)(row);
        float4 b = *(const float4*)(row + 4);
        acc[0] += wt * a.x; acc[1] += wt * a.y; acc[2] += wt * a.z; acc[3] += wt * a.w;
        acc[4] += wt * b.x; acc[5] += wt * b.y; acc[6] += wt * b.z; acc[7] += wt * b.w;
    }

    float ss = 0.f;
#pragma unroll
    for (int d = 0; d < 8; ++d) ss += acc[d] * acc[d];
#pragma unroll
    for (int off = 16; off >= 1; off >>= 1) ss += __shfl_xor(ss, off, 32);
    float scale = qnorm[q] / fmaxf(sqrtf(ss), EPS);

    float4 oa = {acc[0] * scale, acc[1] * scale, acc[2] * scale, acc[3] * scale};
    float4 ob = {acc[4] * scale, acc[5] * scale, acc[6] * scale, acc[7] * scale};
    float* op = out + (size_t)q * D_DIM + lane * 8;
    *(float4*)(op)     = oa;
    *(float4*)(op + 4) = ob;
}

// ---------------------------------------------------------------------------
extern "C" void kernel_launch(void* const* d_in, const int* in_sizes, int n_in,
                              void* d_out, int out_size, void* d_ws, size_t ws_size,
                              hipStream_t stream) {
    const float* query  = (const float*)d_in[0];
    const float* memory = (const float*)d_in[1];
    const int Bq = in_sizes[0] / D_DIM;   // 8192
    const int Mr = in_sizes[1] / D_DIM;   // 32768

    // Workspace carve-up
    char* ws = (char*)d_ws;
    _Float16* nq = (_Float16*)ws;                                   // Bq*D*2
    size_t off = (size_t)Bq * D_DIM * 2;
    _Float16* nm = (_Float16*)(ws + off);                           // Mr*D*2
    off += (size_t)Mr * D_DIM * 2;
    float* qnorm = (float*)(ws + off);                              // Bq*4
    off += (size_t)Bq * 4;
    float* pvals = (float*)(ws + off);                              // Bq*NCHUNK*K*4
    off += (size_t)Bq * NCHUNK * TOPK * 4;
    int* pidx = (int*)(ws + off);                                   // Bq*NCHUNK*K*4

    // 1) normalize rows -> f16 (8 waves/block, one wave per row)
    normalize_rows_kernel<<<(Bq + 7) / 8, 256, 0, stream>>>(query, nq, qnorm, Bq);
    normalize_rows_kernel<<<(Mr + 7) / 8, 256, 0, stream>>>(memory, nm, nullptr, Mr);

    // 2) fused WMMA sim + streaming top-8: 4-wave blocks, 128 q-rows/block
    const int chunkRows = Mr / NCHUNK;
    dim3 grid(Bq / QROWS_PER_BLOCK, NCHUNK);
    simtopk_kernel<<<grid, WAVES * 32, 0, stream>>>(nq, nm, pvals, pidx, chunkRows);

    // 3) merge partials, softmax, gather, renormalize
    finalize_kernel<<<(Bq + 7) / 8, 256, 0, stream>>>(pvals, pidx, memory, qnorm,
                                                      (float*)d_out, Bq);
}
